// GLeMaNet_13812614824220
// MI455X (gfx1250) — matrix operand order repaired and checked by reading, not compiled, not run on hardware
//
#include <hip/hip_runtime.h>
#include <hip/hip_bf16.h>

typedef _Float16 f16;
typedef _Float16 v16h __attribute__((ext_vector_type(16)));
typedef float v8f __attribute__((ext_vector_type(8)));
typedef int vi4 __attribute__((ext_vector_type(4)));

#define B_ 8
#define N_ 1024
#define D_ 128
#define H_ 4
#define L_ 4
#define DFC_ 128
#define HD_ 512           // H_*D_
#define ROWS_ 8192        // B_*N_
#define RH_ 32768         // B_*N_*H_

// ---- async global->LDS staging (gfx1250 ASYNCcnt path) with safe fallback ----
#if __has_builtin(__builtin_amdgcn_global_load_async_to_lds_b128) && \
    __has_builtin(__builtin_amdgcn_s_wait_asynccnt)
#define USE_ASYNC_LDS 1
#else
#define USE_ASYNC_LDS 0
#endif

__device__ __forceinline__ void copy16_g2l(const f16* g, f16* l) {
#if USE_ASYNC_LDS
  __builtin_amdgcn_global_load_async_to_lds_b128(
      (__attribute__((address_space(1))) vi4*)g,
      (__attribute__((address_space(3))) vi4*)l, 0, 0);
#else
  *(uint4*)l = *(const uint4*)g;
#endif
}
__device__ __forceinline__ void staging_wait() {
#if USE_ASYNC_LDS
  __builtin_amdgcn_s_wait_asynccnt(0);
#endif
}

// ---------------- f32 -> f16 convert (x4 vectorized) ----------------
__global__ void k_cvt(const float* __restrict__ s, f16* __restrict__ d, int n) {
  int i = (blockIdx.x * blockDim.x + threadIdx.x) * 4;
  if (i >= n) return;
  float4 v = *(const float4*)(s + i);
  union { f16 h[4]; uint2 u; } o;
  o.h[0] = (f16)v.x; o.h[1] = (f16)v.y; o.h[2] = (f16)v.z; o.h[3] = (f16)v.w;
  *(uint2*)(d + i) = o.u;
}

// ---------------- generic batched WMMA GEMM ----------------
// C[M,Nc] = A[M,K] * (TRANSB ? B[Nc,K]^T : B[K,Nc])  (+ bias[col])
// block tile 128x64, 8 waves (4m x 2n), each wave -> 32x32 (4 accumulators,
// 4 WMMAs per K-step, 8 LDS b128 fragment loads -> 1.5x denser than 16x32).
// LDS B is fragment-major: Bs[n][k] so each lane's frag is two 8-half runs (k and k+16).
template <bool TRANSB>
__global__ __launch_bounds__(256) void k_gemm(
    const f16* __restrict__ A, const f16* __restrict__ Bm, float* __restrict__ C,
    const float* __restrict__ bias, int M, int Nc, int K, int lda, int ldb, int ldc,
    long long sAhi, long long sAlo, int iA,
    long long sBhi, long long sBlo, int iB,
    long long sChi, long long sClo, int iC) {
  __shared__ f16 As[128 * 40];  // [m][k], stride 40
  __shared__ f16 Bs[64 * 40];   // [n][k], stride 40 (fragment-major)
  int z = blockIdx.z;
  A  += (long long)(z / iA) * sAhi + (long long)(z % iA) * sAlo;
  Bm += (long long)(z / iB) * sBhi + (long long)(z % iB) * sBlo;
  C  += (long long)(z / iC) * sChi + (long long)(z % iC) * sClo;
  int t = threadIdx.x;
  int lane = t & 31, wid = t >> 5;
  int wm = wid & 3, wn = wid >> 2;
  int m0 = blockIdx.y * 128, n0 = blockIdx.x * 64;
  int hi = lane >> 4;
  v8f acc00 = {}, acc01 = {}, acc10 = {}, acc11 = {};

  for (int k0 = 0; k0 < K; k0 += 32) {
    {  // A tile 128x32: two 16B copies per thread
      int row = t >> 1, seg = (t & 1) * 16;
      const f16* src = A + (size_t)(m0 + row) * lda + k0 + seg;
      copy16_g2l(src, &As[row * 40 + seg]);
      copy16_g2l(src + 8, &As[row * 40 + seg + 8]);
      __builtin_prefetch(src + 32, 0, 3);  // speculative: OOB prefetch is dropped
    }
    if (TRANSB) {  // B global [Nc,K]: already n-major -> straight 16B copy
      int nr = t >> 2, seg = t & 3;
      const f16* src = Bm + (size_t)(n0 + nr) * ldb + k0 + seg * 8;
      copy16_g2l(src, &Bs[nr * 40 + seg * 8]);
      __builtin_prefetch(src + 32, 0, 3);
    } else {       // B global [K,Nc]: vector load along n, scatter-transpose
      int kr = t >> 3, seg = t & 7;
      union { uint4 q; f16 h[8]; } tmp;
      const f16* src = Bm + (size_t)(k0 + kr) * ldb + n0 + seg * 8;
      tmp.q = *(const uint4*)src;
      __builtin_prefetch(src + 32 * ldb, 0, 3);
#pragma unroll
      for (int u = 0; u < 8; ++u) Bs[(seg * 8 + u) * 40 + kr] = tmp.h[u];
    }
    staging_wait();
    __syncthreads();

    // A fragments: lane<16 -> K{0..7,16..23}, lane>=16 -> K{8..15,24..31}
    union { v16h v; uint4 q[2]; } a0, a1, b0, b1;
    int rl0 = wm * 32 + (lane & 15);
    int rl1 = rl0 + 16;
    a0.q[0] = *(const uint4*)&As[rl0 * 40 + (hi ? 8 : 0)];
    a0.q[1] = *(const uint4*)&As[rl0 * 40 + (hi ? 24 : 16)];
    a1.q[0] = *(const uint4*)&As[rl1 * 40 + (hi ? 8 : 0)];
    a1.q[1] = *(const uint4*)&As[rl1 * 40 + (hi ? 24 : 16)];
    // B fragments: lane<16 -> K{0..7,16..23}, lane>=16 -> K{8..15,24..31}
    int kbase = hi ? 8 : 0;
    int col0 = wn * 32 + (lane & 15);
    b0.q[0] = *(const uint4*)&Bs[col0 * 40 + kbase];
    b0.q[1] = *(const uint4*)&Bs[col0 * 40 + kbase + 16];
    b1.q[0] = *(const uint4*)&Bs[(col0 + 16) * 40 + kbase];
    b1.q[1] = *(const uint4*)&Bs[(col0 + 16) * 40 + kbase + 16];

    acc00 = __builtin_amdgcn_wmma_f32_16x16x32_f16(false, a0.v, false, b0.v,
                                                   (short)0, acc00, false, false);
    acc01 = __builtin_amdgcn_wmma_f32_16x16x32_f16(false, a0.v, false, b1.v,
                                                   (short)0, acc01, false, false);
    acc10 = __builtin_amdgcn_wmma_f32_16x16x32_f16(false, a1.v, false, b0.v,
                                                   (short)0, acc10, false, false);
    acc11 = __builtin_amdgcn_wmma_f32_16x16x32_f16(false, a1.v, false, b1.v,
                                                   (short)0, acc11, false, false);
    __syncthreads();
  }

  int c0 = n0 + wn * 32 + (lane & 15);
  int c1 = c0 + 16;
  int r0 = m0 + wm * 32 + (hi << 3);
  int r1 = r0 + 16;
  float bv0 = bias ? bias[c0] : 0.f;
  float bv1 = bias ? bias[c1] : 0.f;
#pragma unroll
  for (int r = 0; r < 8; ++r) {
    C[(size_t)(r0 + r) * ldc + c0] = acc00[r] + bv0;
    C[(size_t)(r0 + r) * ldc + c1] = acc01[r] + bv1;
    C[(size_t)(r1 + r) * ldc + c0] = acc10[r] + bv0;
    C[(size_t)(r1 + r) * ldc + c1] = acc11[r] + bv1;
  }
}

// ---------------- E += E^T in place (coalesced via LDS transpose) ----------------
__global__ __launch_bounds__(256) void k_sym(float* __restrict__ E) {
  int bx = blockIdx.x, by = blockIdx.y;
  if (bx < by) return;  // only upper-triangular tile pairs
  __shared__ float Tb[32][33];
  __shared__ float Ts[32][33];
  float* Ez = E + (size_t)blockIdx.z * N_ * N_;
  int tx = threadIdx.x & 31, ty = threadIdx.x >> 5;  // 32 x 8
  for (int r = ty; r < 32; r += 8)
    Tb[r][tx] = Ez[(size_t)(bx * 32 + r) * N_ + by * 32 + tx];
  __syncthreads();
  for (int r = ty; r < 32; r += 8) {
    float s = Ez[(size_t)(by * 32 + r) * N_ + bx * 32 + tx] + Tb[tx][r];
    Ts[r][tx] = s;
    Ez[(size_t)(by * 32 + r) * N_ + bx * 32 + tx] = s;
  }
  __syncthreads();
  if (bx == by) return;  // diagonal tile already symmetric-updated
  for (int r = ty; r < 32; r += 8)
    Ez[(size_t)(bx * 32 + r) * N_ + by * 32 + tx] = Ts[tx][r];
}

// ------- column softmax of (E*mask) over rows, then *adj, f16 out -------
// masked entries contribute exp(0)=1 to the denominator (reference quirk).
__global__ __launch_bounds__(256) void k_softmax(const float* __restrict__ E,
                                                 const float* __restrict__ adj,
                                                 f16* __restrict__ att) {
  __shared__ float smax[8][32], ssum[8][32];
  int z = blockIdx.y, b = z >> 2;
  const float* Ez = E + (size_t)z * N_ * N_;
  const float* Ab = adj + (size_t)b * N_ * N_;
  f16* Az = att + (size_t)z * N_ * N_;
  int qx = threadIdx.x & 31, pr = threadIdx.x >> 5;
  int q = blockIdx.x * 32 + qx;
  float m = -3.0e38f, s = 0.f;
  for (int p = pr; p < N_; p += 8) {
    float av = Ab[(size_t)p * N_ + q];
    float v = (av > 0.f) ? Ez[(size_t)p * N_ + q] : 0.f;
    float mn = fmaxf(m, v);
    s = s * __expf(m - mn) + __expf(v - mn);
    m = mn;
  }
  smax[pr][qx] = m; ssum[pr][qx] = s;
  __syncthreads();
  if (pr == 0) {
    float mm = m, sv = s;
    for (int i = 1; i < 8; ++i) {
      float m2 = smax[i][qx], s2 = ssum[i][qx];
      float mn = fmaxf(mm, m2);
      sv = sv * __expf(mm - mn) + s2 * __expf(m2 - mn);
      mm = mn;
    }
    smax[0][qx] = mm; ssum[0][qx] = sv;
  }
  __syncthreads();
  float mm = smax[0][qx];
  float inv = 1.f / ssum[0][qx];
  for (int p = pr; p < N_; p += 8) {
    float av = Ab[(size_t)p * N_ + q];
    float o = (av > 0.f) ? __expf(Ez[(size_t)p * N_ + q] - mm) * inv * av : 0.f;
    Az[(size_t)p * N_ + q] = (f16)o;
  }
}

// ---------------- beta = sigmoid([h, relu(az)] . Wb + bb) ----------------
__global__ __launch_bounds__(256) void k_beta(const float* __restrict__ h,
                                              const float* __restrict__ az,
                                              const float* __restrict__ Wb,
                                              const float* __restrict__ bb,
                                              float* __restrict__ beta) {
  int r = blockIdx.x * 8 + (threadIdx.x >> 5);
  int lane = threadIdx.x & 31;
  const float* hr = h + (size_t)r * D_;
  const float* ar = az + (size_t)r * D_;
  float s = 0.f;
  for (int d = lane; d < D_; d += 32)
    s += hr[d] * Wb[d] + fmaxf(ar[d], 0.f) * Wb[D_ + d];
  for (int off = 16; off > 0; off >>= 1) s += __shfl_down(s, off, 32);
  if (lane == 0) beta[r] = 1.f / (1.f + __expf(-(s + bb[0])));
}

// -------- z = beta*h + (1-beta)*relu(az), + f16 copy (x4 vectorized) --------
__global__ void k_zupd(const float* __restrict__ az, const float* __restrict__ h,
                       const float* __restrict__ beta, float* __restrict__ z,
                       f16* __restrict__ zh, int n) {
  int i = (blockIdx.x * blockDim.x + threadIdx.x) * 4;
  if (i >= n) return;
  float bt = beta[i >> 7];
  float4 a = *(const float4*)(az + i);
  float4 hv = *(const float4*)(h + i);
  float4 v;
  v.x = bt * hv.x + (1.f - bt) * fmaxf(a.x, 0.f);
  v.y = bt * hv.y + (1.f - bt) * fmaxf(a.y, 0.f);
  v.z = bt * hv.z + (1.f - bt) * fmaxf(a.z, 0.f);
  v.w = bt * hv.w + (1.f - bt) * fmaxf(a.w, 0.f);
  *(float4*)(z + i) = v;
  union { f16 h4[4]; uint2 u; } o;
  o.h4[0] = (f16)v.x; o.h4[1] = (f16)v.y; o.h4[2] = (f16)v.z; o.h4[3] = (f16)v.w;
  *(uint2*)(zh + i) = o.u;
}

// ---------------- mean over heads (x4 vectorized) ----------------
__global__ void k_headmean(const float* __restrict__ z, float* __restrict__ out, int n) {
  int i = (blockIdx.x * blockDim.x + threadIdx.x) * 4;
  if (i >= n) return;
  int bn = i >> 7, d = i & 127;
  const float* zr = z + (size_t)bn * HD_ + d;
  float4 a = *(const float4*)(zr);
  float4 b = *(const float4*)(zr + 128);
  float4 c = *(const float4*)(zr + 256);
  float4 e = *(const float4*)(zr + 384);
  float4 o;
  o.x = 0.25f * (a.x + b.x + c.x + e.x);
  o.y = 0.25f * (a.y + b.y + c.y + e.y);
  o.z = 0.25f * (a.z + b.z + c.z + e.z);
  o.w = 0.25f * (a.w + b.w + c.w + e.w);
  *(float4*)(out + i) = o;
}

// ---------------- c = c2 - c1, + f16 copy (x4 vectorized) ----------------
__global__ void k_subcvt(const float* __restrict__ c2, const float* __restrict__ c1,
                         float* __restrict__ c, f16* __restrict__ ch, int n) {
  int i = (blockIdx.x * blockDim.x + threadIdx.x) * 4;
  if (i >= n) return;
  float4 a = *(const float4*)(c2 + i);
  float4 b = *(const float4*)(c1 + i);
  float4 v;
  v.x = a.x - b.x; v.y = a.y - b.y; v.z = a.z - b.z; v.w = a.w - b.w;
  *(float4*)(c + i) = v;
  union { f16 h4[4]; uint2 u; } o;
  o.h4[0] = (f16)v.x; o.h4[1] = (f16)v.y; o.h4[2] = (f16)v.z; o.h4[3] = (f16)v.w;
  *(uint2*)(ch + i) = o.u;
}

// ---------------- masked mean pool + 4-layer MLP head ----------------
__global__ __launch_bounds__(128) void k_final(
    const float* __restrict__ c, const float* __restrict__ valid,
    const float* __restrict__ W0, const float* __restrict__ b0,
    const float* __restrict__ Wm, const float* __restrict__ bm,
    const float* __restrict__ Wl, const float* __restrict__ bl,
    float* __restrict__ out) {
  __shared__ float pool[DFC_], buf[DFC_], red[DFC_];
  int b = blockIdx.x, t = threadIdx.x;
  const float* cb = c + (size_t)b * N_ * D_;
  const float* vb = valid + (size_t)b * N_;
  float s = 0.f, vs = 0.f;
  for (int n = 0; n < N_; ++n) s += cb[(size_t)n * D_ + t] * vb[n];
  for (int n = t; n < N_; n += DFC_) vs += vb[n];
  red[t] = vs;
  __syncthreads();
  for (int off = 64; off > 0; off >>= 1) {
    if (t < off) red[t] += red[t + off];
    __syncthreads();
  }
  pool[t] = s / red[0];
  __syncthreads();
  float y = b0[t];
  for (int d = 0; d < DFC_; ++d) y += pool[d] * W0[d * DFC_ + t];
  buf[t] = fmaxf(y, 0.f);
  __syncthreads();
  for (int m = 0; m < 2; ++m) {
    float yy = bm[m * DFC_ + t];
    const float* W = Wm + (size_t)m * DFC_ * DFC_;
    for (int d = 0; d < DFC_; ++d) yy += buf[d] * W[d * DFC_ + t];
    __syncthreads();
    buf[t] = fmaxf(yy, 0.f);
    __syncthreads();
  }
  red[t] = buf[t] * Wl[t];
  __syncthreads();
  for (int off = 64; off > 0; off >>= 1) {
    if (t < off) red[t] += red[t + off];
    __syncthreads();
  }
  if (t == 0) out[b] = 1.f / (1.f + __expf(-(red[0] + bl[0])));
}

extern "C" void kernel_launch(void* const* d_in, const int* in_sizes, int n_in,
                              void* d_out, int out_size, void* d_ws, size_t ws_size,
                              hipStream_t stream) {
  const float* x      = (const float*)d_in[0];
  const float* adj1   = (const float*)d_in[1];
  const float* adj2   = (const float*)d_in[2];
  const float* valid  = (const float*)d_in[3];
  const float* Wemb   = (const float*)d_in[4];
  const float* Wh     = (const float*)d_in[5];
  const float* bh     = (const float*)d_in[6];
  const float* We     = (const float*)d_in[7];
  const float* Wb     = (const float*)d_in[8];
  const float* bbp    = (const float*)d_in[9];
  const float* Wfc0   = (const float*)d_in[10];
  const float* bfc0   = (const float*)d_in[11];
  const float* Wfcm   = (const float*)d_in[12];
  const float* bfcm   = (const float*)d_in[13];
  const float* Wfcl   = (const float*)d_in[14];
  const float* bfcl   = (const float*)d_in[15];
  (void)in_sizes; (void)n_in; (void)out_size; (void)ws_size;

  char* wp = (char*)d_ws;
  auto alloc = [&](size_t bytes) -> char* {
    char* p = wp;
    wp += (bytes + 255) & ~(size_t)255;
    return p;
  };
  f16*   xH    = (f16*)alloc((size_t)ROWS_ * D_ * 2);
  f16*   WembH = (f16*)alloc((size_t)D_ * D_ * 2);
  f16*   WhH   = (f16*)alloc((size_t)L_ * D_ * HD_ * 2);
  f16*   WeH   = (f16*)alloc((size_t)L_ * D_ * D_ * 2);
  float* cF    = (float*)alloc((size_t)ROWS_ * D_ * 4);
  f16*   cH    = (f16*)alloc((size_t)ROWS_ * D_ * 2);
  float* hF    = (float*)alloc((size_t)ROWS_ * HD_ * 4);
  f16*   hH    = (f16*)alloc((size_t)ROWS_ * HD_ * 2);
  float* gF    = (float*)alloc((size_t)ROWS_ * HD_ * 4);
  f16*   gH    = (f16*)alloc((size_t)ROWS_ * HD_ * 2);
  float* EF    = (float*)alloc((size_t)32 * N_ * N_ * 4);
  f16*   attH  = (f16*)alloc((size_t)32 * N_ * N_ * 2);
  float* azF   = (float*)alloc((size_t)ROWS_ * HD_ * 4);
  float* zF    = (float*)alloc((size_t)ROWS_ * HD_ * 4);
  f16*   zH    = (f16*)alloc((size_t)ROWS_ * HD_ * 2);
  float* betaB = (float*)alloc((size_t)RH_ * 4);
  float* c1F   = (float*)alloc((size_t)ROWS_ * D_ * 4);
  float* c2F   = (float*)alloc((size_t)ROWS_ * D_ * 4);

  auto cvt = [&](const float* s, f16* d, int n) {
    k_cvt<<<(n / 4 + 255) / 256, 256, 0, stream>>>(s, d, n);
  };
  auto gemm = [&](bool tb, const f16* A, const f16* Bm, float* C, const float* bias,
                  int M, int Nc, int K, int lda, int ldb, int ldc, int batch,
                  long long sAhi, long long sAlo, int iA,
                  long long sBhi, long long sBlo, int iB,
                  long long sChi, long long sClo, int iC) {
    dim3 g(Nc / 64, M / 128, batch), blk(256);
    if (tb)
      k_gemm<true><<<g, blk, 0, stream>>>(A, Bm, C, bias, M, Nc, K, lda, ldb, ldc,
                                          sAhi, sAlo, iA, sBhi, sBlo, iB, sChi, sClo, iC);
    else
      k_gemm<false><<<g, blk, 0, stream>>>(A, Bm, C, bias, M, Nc, K, lda, ldb, ldc,
                                           sAhi, sAlo, iA, sBhi, sBlo, iB, sChi, sClo, iC);
  };

  // --- weight / input conversions ---
  cvt(x, xH, ROWS_ * D_);
  cvt(Wemb, WembH, D_ * D_);
  cvt(Wh, WhH, L_ * D_ * HD_);
  cvt(We, WeH, L_ * D_ * D_);

  // --- embedding: c = x @ W_embede ---
  gemm(false, xH, WembH, cF, nullptr, ROWS_, D_, D_, D_, D_, D_, 1,
       0, 0, 1, 0, 0, 1, 0, 0, 1);
  cvt(cF, cH, ROWS_ * D_);

  const long long sBN = (long long)N_ * HD_;  // per-b stride in [B,N,H,D] flats
  for (int k = 0; k < L_; ++k) {
    // h = c @ Wh[k] + bh[k]
    gemm(false, cH, WhH + (size_t)k * D_ * HD_, hF, bh + (size_t)k * HD_,
         ROWS_, HD_, D_, D_, HD_, HD_, 1, 0, 0, 1, 0, 0, 1, 0, 0, 1);
    cvt(hF, hH, ROWS_ * HD_);
    // g[b,n,i,:] = h[b,n,i,:] @ We[k]  (flatten to [B*N*H, D])
    gemm(false, hH, WeH + (size_t)k * D_ * D_, gF, nullptr,
         RH_, D_, D_, D_, D_, D_, 1, 0, 0, 1, 0, 0, 1, 0, 0, 1);
    cvt(gF, gH, ROWS_ * HD_);
    // E[z] = G[b,:,i,:] @ H[b,:,i,:]^T  (z = b*4+i, 32 batches, NT GEMM)
    gemm(true, gH, hH, EF, nullptr, N_, N_, D_, HD_, HD_, N_, 32,
         sBN, D_, H_, sBN, D_, H_, (long long)N_ * N_, 0, 1);
    k_sym<<<dim3(32, 32, 32), 256, 0, stream>>>(EF);

    int nhop = k + 1;
    for (int br = 0; br < 2; ++br) {
      const float* adj = (br == 0) ? adj1 : adj2;
      float* cOut = (br == 0) ? c1F : c2F;
      k_softmax<<<dim3(N_ / 32, 32), 256, 0, stream>>>(EF, adj, attH);
      // az0 = att @ h  (z = h initially)
      gemm(false, attH, hH, azF, nullptr, N_, D_, N_, N_, HD_, HD_, 32,
           (long long)N_ * N_, 0, 1, sBN, D_, H_, sBN, D_, H_);
      k_beta<<<RH_ / 8, 256, 0, stream>>>(hF, azF, Wb + (size_t)k * 2 * D_,
                                          bbp + k, betaB);
      k_zupd<<<(RH_ * D_ / 4 + 255) / 256, 256, 0, stream>>>(azF, hF, betaB, zF, zH,
                                                             RH_ * D_);
      for (int hop = 1; hop < nhop; ++hop) {
        gemm(false, attH, zH, azF, nullptr, N_, D_, N_, N_, HD_, HD_, 32,
             (long long)N_ * N_, 0, 1, sBN, D_, H_, sBN, D_, H_);
        k_zupd<<<(RH_ * D_ / 4 + 255) / 256, 256, 0, stream>>>(azF, hF, betaB, zF, zH,
                                                               RH_ * D_);
      }
      k_headmean<<<(ROWS_ * D_ / 4 + 255) / 256, 256, 0, stream>>>(zF, cOut,
                                                                   ROWS_ * D_);
    }
    k_subcvt<<<(ROWS_ * D_ / 4 + 255) / 256, 256, 0, stream>>>(c2F, c1F, cF, cH,
                                                               ROWS_ * D_);
  }

  k_final<<<B_, 128, 0, stream>>>(cF, valid, Wfc0, bfc0, Wfcm, bfcm, Wfcl, bfcl,
                                  (float*)d_out);
}
